// LocalGNN_16363825397775
// MI455X (gfx1250) — matrix operand order repaired and checked by reading, not compiled, not run on hardware
//
#include <hip/hip_runtime.h>
#include <hip/hip_bf16.h>

typedef __attribute__((ext_vector_type(16))) _Float16 v16h;
typedef __attribute__((ext_vector_type(8)))  float    v8f;

#define N_NODES   10000
#define N_EDGES   320000
#define HIDC      128
#define NHEAD     4
#define NLAYER    2
#define NEG_SLOPE 0.2f
#define BN_EPS    1e-5f
#define NFLAT     (N_EDGES + N_NODES)

// ---------------------------------------------------------------------------
// Pack f32 weight [K x Ncols] into fragment-major f16 for WMMA B operand:
//   P[((kc*(Ncols/16) + tile)*32 + lane)*16 + e] =
//       W[(kc*32 + (lane>>4)*8 + (e&7) + (e>>3)*16) * Ncols + tile*16 + (lane&15)]
// so each lane's 16-half B fragment is 32 contiguous bytes.
// ---------------------------------------------------------------------------
__global__ void k_pack_b(const float* __restrict__ W, _Float16* __restrict__ P,
                         int K, int Ncols) {
  int i = blockIdx.x * blockDim.x + threadIdx.x;
  if (i >= K * Ncols) return;
  int e    = i & 15;
  int lane = (i >> 4) & 31;
  int rest = i >> 9;
  int nt   = Ncols >> 4;
  int tile = rest % nt;
  int kc   = rest / nt;
  int k = kc * 32 + ((lane >> 4) << 3) + (e & 7) + ((e >> 3) << 4);
  int n = tile * 16 + (lane & 15);
  P[i] = (_Float16)W[(size_t)k * Ncols + n];
}

// ---------------------------------------------------------------------------
// WMMA GEMM:  C[M x Ncols] = A_f32[M x K] * Bpacked_f16[K x Ncols] (+bias)
// One wave -> 16 x (16*NT) tile. A fragment: four float4 loads + cvt;
// B fragment: one contiguous 32B v16h load (fragment-major packed).
// ---------------------------------------------------------------------------
template <int K, int NT>
__global__ void k_gemm_wmma(const float* __restrict__ A,
                            const _Float16* __restrict__ B,
                            const float* __restrict__ bias,
                            float* __restrict__ C, int M, int Ncols) {
  const int wave = threadIdx.x >> 5;
  const int lane = threadIdx.x & 31;
  const int row0 = (blockIdx.x * (blockDim.x >> 5) + wave) * 16;
  if (row0 >= M) return;                 // wave-uniform: EXEC stays all-ones
  const int col0 = blockIdx.y * (16 * NT);
  const int ntiles = Ncols >> 4;
  const int tile0  = col0 >> 4;

  v8f acc[NT] = {};

  const int mA  = row0 + (lane & 15);
  const int kbA = (lane >> 4) << 3;      // A half: K offset 0 or 8

#pragma unroll
  for (int k0 = 0; k0 < K; k0 += 32) {
    const int kc = k0 >> 5;
    // ---- A fragment: 4x float4 (32B-aligned), convert to f16 ----
    const float4* ap4 = (const float4*)(A + (size_t)mA * K + k0 + kbA);
    float4 a0 = ap4[0];
    float4 a1 = ap4[1];        // K = kb + 0..7
    float4 a2 = ap4[4];
    float4 a3 = ap4[5];        // K = 16 + kb + 0..7
    v16h a;
    a[0]  = (_Float16)a0.x; a[1]  = (_Float16)a0.y; a[2]  = (_Float16)a0.z; a[3]  = (_Float16)a0.w;
    a[4]  = (_Float16)a1.x; a[5]  = (_Float16)a1.y; a[6]  = (_Float16)a1.z; a[7]  = (_Float16)a1.w;
    a[8]  = (_Float16)a2.x; a[9]  = (_Float16)a2.y; a[10] = (_Float16)a2.z; a[11] = (_Float16)a2.w;
    a[12] = (_Float16)a3.x; a[13] = (_Float16)a3.y; a[14] = (_Float16)a3.z; a[15] = (_Float16)a3.w;

#pragma unroll
    for (int t = 0; t < NT; ++t) {
      // ---- B fragment: contiguous 32 bytes per lane ----
      const v16h* bp = (const v16h*)(B + ((size_t)(kc * ntiles + tile0 + t) * 32 + lane) * 16);
      v16h b = *bp;
      acc[t] = __builtin_amdgcn_wmma_f32_16x16x32_f16(
          false, a, false, b, (short)0, acc[t], false, false);
    }
  }

  const int mb = row0 + ((lane >> 4) << 3);
  const int nB = lane & 15;
#pragma unroll
  for (int t = 0; t < NT; ++t) {
    const int n = col0 + 16 * t + nB;
    const float bv = bias ? bias[n] : 0.0f;
#pragma unroll
    for (int r = 0; r < 8; ++r)
      C[(size_t)(mb + r) * Ncols + n] = acc[t][r] + bv;
  }
}

// ---------------------------------------------------------------------------
__global__ void k_fill(float* __restrict__ p, float v, int n) {
  int i = blockIdx.x * blockDim.x + threadIdx.x;
  if (i < n) p[i] = v;
}

// loop_attr accumulation: rows E..E+N of ea_f get sum of incoming edge feats
__global__ void k_loop_accum(const float* __restrict__ ea, const int* __restrict__ dst,
                             float* __restrict__ loop_rows, float* __restrict__ deg) {
  unsigned int i = blockIdx.x * blockDim.x + threadIdx.x;
  if (i >= (unsigned)N_EDGES * HIDC) return;
  int e = i >> 7, c = i & 127;
  int d = dst[e];
  atomicAdd(&loop_rows[(size_t)d * HIDC + c], ea[(size_t)e * HIDC + c]);
  if (c == 0) atomicAdd(&deg[d], 1.0f);
}

__global__ void k_loop_div(float* __restrict__ loop_rows, const float* __restrict__ deg) {
  int i = blockIdx.x * blockDim.x + threadIdx.x;
  if (i >= N_NODES * HIDC) return;
  loop_rows[i] /= fmaxf(deg[i >> 7], 1.0f);
}

// V[k][h] = sum_c W[k, h*128+c] * att[h, c]   (tiny: 128x4 result)
__global__ void k_attvec(const float* __restrict__ Wl, const float* __restrict__ attl,
                         float* __restrict__ V) {
  int i = blockIdx.x * blockDim.x + threadIdx.x;
  if (i >= HIDC * NHEAD) return;
  int k = i >> 2, hd = i & 3;
  float s = 0.f;
  for (int c = 0; c < HIDC; ++c)
    s += Wl[(size_t)k * (NHEAD * HIDC) + hd * HIDC + c] * attl[hd * HIDC + c];
  V[k * NHEAD + hd] = s;
}

// out[r][h] = X[r,:] . V[:,h]  for h=0..3
__global__ void k_row_dot4(const float* __restrict__ X, const float* __restrict__ V,
                           float* __restrict__ out, int rows) {
  int n = blockIdx.x * blockDim.x + threadIdx.x;
  if (n >= rows) return;
  const float* xr = X + (size_t)n * HIDC;
  float s0 = 0, s1 = 0, s2 = 0, s3 = 0;
  for (int k = 0; k < HIDC; ++k) {
    float x = xr[k];
    s0 += x * V[k * 4 + 0]; s1 += x * V[k * 4 + 1];
    s2 += x * V[k * 4 + 2]; s3 += x * V[k * 4 + 3];
  }
  float* o = out + (size_t)n * 4;
  o[0] = s0; o[1] = s1; o[2] = s2; o[3] = s3;
}

__device__ inline void atomicMaxFloat(float* addr, float val) {
  // monotone bit trick; addr initialized to -inf (0xFF800000)
  if (val >= 0.f) atomicMax((int*)addr, __float_as_int(val));
  else            atomicMin((unsigned int*)addr, (unsigned int)__float_as_int(val));
}

// logits + leaky_relu + segment max
__global__ void k_alpha1(const float* __restrict__ a_src, const float* __restrict__ a_dst,
                         const float* __restrict__ a_e, const int* __restrict__ src,
                         const int* __restrict__ dst, float* __restrict__ logit,
                         float* __restrict__ amax) {
  int i = blockIdx.x * blockDim.x + threadIdx.x;
  if (i >= NFLAT) return;
  int s = (i < N_EDGES) ? src[i] : (i - N_EDGES);
  int d = (i < N_EDGES) ? dst[i] : (i - N_EDGES);
#pragma unroll
  for (int hd = 0; hd < NHEAD; ++hd) {
    float v = a_src[s * 4 + hd] + a_dst[d * 4 + hd] + a_e[(size_t)i * 4 + hd];
    v = (v >= 0.f) ? v : NEG_SLOPE * v;
    logit[(size_t)i * 4 + hd] = v;
    atomicMaxFloat(&amax[d * 4 + hd], v);
  }
}

// exp(logit - max) + segment sum (overwrites logit with alpha)
__global__ void k_alpha2(const int* __restrict__ dst, float* __restrict__ logit,
                         const float* __restrict__ amax, float* __restrict__ denom) {
  int i = blockIdx.x * blockDim.x + threadIdx.x;
  if (i >= NFLAT) return;
  int d = (i < N_EDGES) ? dst[i] : (i - N_EDGES);
#pragma unroll
  for (int hd = 0; hd < NHEAD; ++hd) {
    float a = __expf(logit[(size_t)i * 4 + hd] - amax[d * 4 + hd]);
    logit[(size_t)i * 4 + hd] = a;
    atomicAdd(&denom[d * 4 + hd], a);
  }
}

// out[d, c] += (1/H) * sum_h alpha_norm[i,h] * xh[s, h*128 + c]
// xh (20.5 MB) is L2-resident -> gather served from L2, not HBM.
__global__ void k_message(const int* __restrict__ src, const int* __restrict__ dst,
                          const float* __restrict__ alpha, const float* __restrict__ denom,
                          const float* __restrict__ xh, float* __restrict__ out) {
  unsigned int t = blockIdx.x * blockDim.x + threadIdx.x;
  if (t >= (unsigned)NFLAT * HIDC) return;
  int i = t >> 7, c = t & 127;
  int s = (i < N_EDGES) ? src[i] : (i - N_EDGES);
  int d = (i < N_EDGES) ? dst[i] : (i - N_EDGES);
  float acc = 0.f;
#pragma unroll
  for (int hd = 0; hd < NHEAD; ++hd) {
    float w = alpha[(size_t)i * 4 + hd] / (denom[d * 4 + hd] + 1e-16f);
    acc += w * xh[(size_t)s * (NHEAD * HIDC) + hd * HIDC + c];
  }
  atomicAdd(&out[(size_t)d * HIDC + c], acc * (1.0f / NHEAD));
}

// per-channel mean / rsqrt(var+eps) over N (bias[l] cancels inside BN exactly)
__global__ void k_bnstats(const float* __restrict__ out, float* __restrict__ mu,
                          float* __restrict__ rsig) {
  int c = blockIdx.x;
  float s = 0.f, s2 = 0.f;
  for (int n = threadIdx.x; n < N_NODES; n += blockDim.x) {
    float v = out[(size_t)n * HIDC + c];
    s += v; s2 += v * v;
  }
  __shared__ float sh[256], sh2[256];
  sh[threadIdx.x] = s; sh2[threadIdx.x] = s2;
  __syncthreads();
  for (int st = 128; st > 0; st >>= 1) {
    if ((int)threadIdx.x < st) {
      sh[threadIdx.x] += sh[threadIdx.x + st];
      sh2[threadIdx.x] += sh2[threadIdx.x + st];
    }
    __syncthreads();
  }
  if (threadIdx.x == 0) {
    float m = sh[0] * (1.0f / N_NODES);
    float var = sh2[0] * (1.0f / N_NODES) - m * m;
    mu[c] = m;
    rsig[c] = rsqrtf(var + BN_EPS);
  }
}

// h = relu(BN(out)) + h ; optionally mirror to dest (final output)
__global__ void k_bn_apply(const float* __restrict__ out, const float* __restrict__ mu,
                           const float* __restrict__ rsig, const float* __restrict__ gamma,
                           const float* __restrict__ beta, float* __restrict__ h,
                           float* __restrict__ dest) {
  int i = blockIdx.x * blockDim.x + threadIdx.x;
  if (i >= N_NODES * HIDC) return;
  int c = i & 127;
  float v = (out[i] - mu[c]) * rsig[c] * gamma[c] + beta[c];
  v = fmaxf(v, 0.f) + h[i];
  h[i] = v;
  if (dest) dest[i] = v;
}

// ---------------------------------------------------------------------------
extern "C" void kernel_launch(void* const* d_in, const int* in_sizes, int n_in,
                              void* d_out, int out_size, void* d_ws, size_t ws_size,
                              hipStream_t stream) {
  const float* x         = (const float*)d_in[0];
  const int*   ei        = (const int*)d_in[1];
  const int*   src       = ei;
  const int*   dst       = ei + N_EDGES;
  const float* edge_attr = (const float*)d_in[2];
  const float* enc_W     = (const float*)d_in[4];
  const float* enc_b     = (const float*)d_in[5];
  const float* eenc_W    = (const float*)d_in[6];
  const float* eenc_b    = (const float*)d_in[7];
  const float* W         = (const float*)d_in[8];   // [2,128,512]
  const float* We        = (const float*)d_in[9];   // [2,128,512]
  const float* att_src   = (const float*)d_in[10];  // [2,4,128]
  const float* att_dst   = (const float*)d_in[11];
  const float* att_edge  = (const float*)d_in[12];
  const float* gamma     = (const float*)d_in[14];  // [2,128]
  const float* beta      = (const float*)d_in[15];

  // --- workspace carve (256B aligned) ---
  char* wp = (char*)d_ws;
  auto carve = [&](size_t bytes) -> char* {
    char* p = wp; wp += (bytes + 255) & ~(size_t)255; return p;
  };
  float*     ea_f   = (float*)carve((size_t)NFLAT * HIDC * 4);   // [E+N,128]
  float*     h      = (float*)carve((size_t)N_NODES * HIDC * 4); // [N,128]
  float*     xh     = (float*)carve((size_t)N_NODES * NHEAD * HIDC * 4); // [N,512]
  float*     a_eb   = (float*)carve((size_t)NFLAT * 4 * 4);
  float*     alpha  = (float*)carve((size_t)NFLAT * 4 * 4);      // logits -> alpha
  float*     a_srcb = (float*)carve((size_t)N_NODES * 4 * 4);
  float*     a_dstb = (float*)carve((size_t)N_NODES * 4 * 4);
  float*     amax   = (float*)carve((size_t)N_NODES * 4 * 4);
  float*     denom  = (float*)carve((size_t)N_NODES * 4 * 4);
  float*     deg    = (float*)carve((size_t)N_NODES * 4);
  float*     outb   = (float*)carve((size_t)N_NODES * HIDC * 4);
  float*     mu     = (float*)carve(HIDC * 4);
  float*     rsig   = (float*)carve(HIDC * 4);
  float*     Vs     = (float*)carve(HIDC * NHEAD * 4);
  float*     Vd     = (float*)carve(HIDC * NHEAD * 4);
  float*     Ve     = (float*)carve(HIDC * NHEAD * 4);
  _Float16*  encW_p  = (_Float16*)carve(128 * 128 * 2);
  _Float16*  eencW_p = (_Float16*)carve(64 * 128 * 2);
  _Float16*  W_p     = (_Float16*)carve((size_t)NLAYER * 128 * 512 * 2);

  const int TB = 256;
  auto blocks = [&](long long n) { return (int)((n + TB - 1) / TB); };

  // weights -> packed f16 fragments (B operands)
  k_pack_b<<<blocks(128 * 128), TB, 0, stream>>>(enc_W, encW_p, 128, 128);
  k_pack_b<<<blocks(64 * 128), TB, 0, stream>>>(eenc_W, eencW_p, 64, 128);
  for (int l = 0; l < NLAYER; ++l)
    k_pack_b<<<blocks(128 * 512), TB, 0, stream>>>(
        W + (size_t)l * 128 * 512, W_p + (size_t)l * 128 * 512, 128, 512);

  // node encoder: h = x @ enc_W + enc_b      [10000 x 128]
  k_gemm_wmma<128, 4><<<dim3((N_NODES + 63) / 64, 128 / 64), 128, 0, stream>>>(
      x, encW_p, enc_b, h, N_NODES, HIDC);
  // edge encoder: ea = edge_attr @ eenc_W + eenc_b   [320000 x 128]
  k_gemm_wmma<64, 4><<<dim3(N_EDGES / 64, 128 / 64), 128, 0, stream>>>(
      edge_attr, eencW_p, eenc_b, ea_f, N_EDGES, HIDC);

  // self-loop attrs = per-dst mean of incoming edge feats -> rows E..E+N of ea_f
  float* loop_rows = ea_f + (size_t)N_EDGES * HIDC;
  k_fill<<<blocks(N_NODES * HIDC), TB, 0, stream>>>(loop_rows, 0.f, N_NODES * HIDC);
  k_fill<<<blocks(N_NODES), TB, 0, stream>>>(deg, 0.f, N_NODES);
  k_loop_accum<<<blocks((long long)N_EDGES * HIDC), TB, 0, stream>>>(ea_f, dst, loop_rows, deg);
  k_loop_div<<<blocks(N_NODES * HIDC), TB, 0, stream>>>(loop_rows, deg);

  for (int l = 0; l < NLAYER; ++l) {
    const float* Wl  = W + (size_t)l * 128 * 512;
    const float* Wel = We + (size_t)l * 128 * 512;
    k_attvec<<<blocks(HIDC * NHEAD), TB, 0, stream>>>(Wl, att_src + l * 512, Vs);
    k_attvec<<<blocks(HIDC * NHEAD), TB, 0, stream>>>(Wl, att_dst + l * 512, Vd);
    k_attvec<<<blocks(HIDC * NHEAD), TB, 0, stream>>>(Wel, att_edge + l * 512, Ve);

    // xh = h @ W[l]   [10000 x 512]
    k_gemm_wmma<128, 4><<<dim3((N_NODES + 63) / 64, 512 / 64), 128, 0, stream>>>(
        h, W_p + (size_t)l * 128 * 512, nullptr, xh, N_NODES, NHEAD * HIDC);

    // attention scalars
    k_row_dot4<<<blocks(N_NODES), TB, 0, stream>>>(h, Vs, a_srcb, N_NODES);
    k_row_dot4<<<blocks(N_NODES), TB, 0, stream>>>(h, Vd, a_dstb, N_NODES);
    k_row_dot4<<<blocks(NFLAT), TB, 0, stream>>>(ea_f, Ve, a_eb, NFLAT);

    // segment softmax
    k_fill<<<blocks(N_NODES * 4), TB, 0, stream>>>(amax, -__builtin_inff(), N_NODES * 4);
    k_fill<<<blocks(N_NODES * 4), TB, 0, stream>>>(denom, 0.f, N_NODES * 4);
    k_fill<<<blocks(N_NODES * HIDC), TB, 0, stream>>>(outb, 0.f, N_NODES * HIDC);
    k_alpha1<<<blocks(NFLAT), TB, 0, stream>>>(a_srcb, a_dstb, a_eb, src, dst, alpha, amax);
    k_alpha2<<<blocks(NFLAT), TB, 0, stream>>>(dst, alpha, amax, denom);

    // message aggregation (head-fused)
    k_message<<<blocks((long long)NFLAT * HIDC), TB, 0, stream>>>(src, dst, alpha, denom, xh, outb);

    // batch-norm + relu + residual
    k_bnstats<<<HIDC, 256, 0, stream>>>(outb, mu, rsig);
    k_bn_apply<<<blocks(N_NODES * HIDC), TB, 0, stream>>>(
        outb, mu, rsig, gamma + l * HIDC, beta + l * HIDC, h,
        (l == NLAYER - 1) ? (float*)d_out : nullptr);
  }
}
